// BurstDeblurNet_41025527611660
// MI455X (gfx1250) — matrix operand order repaired and checked by reading, not compiled, not run on hardware
//
#include <hip/hip_runtime.h>
#include <cstdint>
#include <cstddef>

// ---------------------------------------------------------------------------
// Types for WMMA / TDM builtins
// ---------------------------------------------------------------------------
typedef __attribute__((ext_vector_type(16))) _Float16 v16h;
typedef __attribute__((ext_vector_type(8)))  _Float16 v8h;
typedef __attribute__((ext_vector_type(8)))  float    v8f;
typedef __attribute__((ext_vector_type(4)))  unsigned int v4u;
typedef __attribute__((ext_vector_type(8)))  int      v8i;
typedef __attribute__((ext_vector_type(4)))  int      v4i;

#if defined(__HIP_DEVICE_COMPILE__)
#if __has_builtin(__builtin_amdgcn_tensor_load_to_lds) && __has_builtin(__builtin_amdgcn_s_wait_tensorcnt)
#define USE_TDM 1
#endif
#endif
#ifndef USE_TDM
#define USE_TDM 0
#endif

#define ACT_NONE 0
#define ACT_RELU 1
#define ACT_SIG  2
#define ACT_GELU 3

// ---------------------------------------------------------------------------
// WMMA fragment helpers (wave32, 16x16x32 f16 -> f32)
//   A 16x32 (MxK): lane L -> M = L%16, element hh -> K = (L/16)*16 + hh
//     rows stored m-major (pitch mult. of 8 halves) -> two ds_load_b128 per frag
//   B 32x16 (KxN): lane L -> N = L%16, element hh -> K = (L/16)*16 + hh
//     tiles stored n-major (s[n][k])               -> two ds_load_b128 per frag
//   C 16x16 f32:   vgpr r -> M = r + 8*(L/16), N = L%16
// ---------------------------------------------------------------------------
__device__ inline v8f wmma_f16(v16h a, v16h b, v8f c) {
  return __builtin_amdgcn_wmma_f32_16x16x32_f16(false, a, false, b, (short)0, c, false, false);
}

__device__ inline v16h load_frag_a(const _Float16* s, int pitch) {
  const int lane = threadIdx.x & 31;
  const int m = lane & 15, hi = lane >> 4;
  v8h lo = *(const v8h*)(s + m * pitch + hi * 16);
  v8h hh = *(const v8h*)(s + m * pitch + hi * 16 + 8);
  v16h a;
#pragma unroll
  for (int i = 0; i < 8; ++i) { a[i] = lo[i]; a[i + 8] = hh[i]; }
  return a;
}

__device__ inline v16h load_frag_bT(const _Float16* s, int pitch) { // B(k,n) at s[n*pitch+k]
  const int lane = threadIdx.x & 31;
  const int n = lane & 15, hi = lane >> 4;
  v8h lo = *(const v8h*)(s + n * pitch + hi * 16);
  v8h hh = *(const v8h*)(s + n * pitch + hi * 16 + 8);
  v16h b;
#pragma unroll
  for (int i = 0; i < 8; ++i) { b[i] = lo[i]; b[i + 8] = hh[i]; }
  return b;
}

__device__ inline float act_apply(float v, int act) {
  if (act == ACT_RELU) return fmaxf(v, 0.f);
  if (act == ACT_SIG)  return 1.f / (1.f + __expf(-v));
  if (act == ACT_GELU) {
    float c = v + 0.044715f * v * v * v;
    return 0.5f * v * (1.f + tanhf(0.7978845608f * c));
  }
  return v;
}

__device__ inline void store_tile(v8f acc, int gm0, int gn0, int M, int N,
                                  const float* bias, const _Float16* resid,
                                  _Float16* outH, float* outF, int act) {
  const int lane = threadIdx.x & 31;
  const int nl = lane & 15, hi = lane >> 4;
#pragma unroll
  for (int r = 0; r < 8; ++r) {
    int m = gm0 + r + hi * 8;
    int n = gn0 + nl;
    if (m < M && n < N) {
      float v = acc[r];
      if (bias)  v += bias[m];
      if (resid) v += (float)resid[(size_t)m * N + n];
      v = act_apply(v, act);
      if (outH) outH[(size_t)m * N + n] = (_Float16)v;
      if (outF) outF[(size_t)m * N + n] = v;
    }
  }
}

// ---------------------------------------------------------------------------
// Generic WMMA GEMM: out[m][n] = act(bias[m] + resid[m][n] + sum_k A[m][k]*B(k,n))
// mode 0: B(k,n) = B1[k*N+n]   (dense, feature-major activations)
// mode 1: B(k,n) = fused 3x3 im2col gather (virtual concat of B1|B2 at C1)
// Block: 64x64 C-tile, 8 waves, 2 16x16 tiles/wave. A-tile via TDM when avail.
// ---------------------------------------------------------------------------
__launch_bounds__(256)
__global__ void k_gemm(const _Float16* __restrict__ A,
                       const _Float16* __restrict__ B1,
                       const _Float16* __restrict__ B2,
                       const float* __restrict__ bias,
                       const _Float16* __restrict__ resid,
                       _Float16* __restrict__ outH,
                       float* __restrict__ outF,
                       int M, int N, int K, int mode, int C1,
                       int Wimg, int Himg, int act) {
  __shared__ __align__(16) _Float16 As[64 * 40];   // A tile m-major, pitch 40 halves (TDM pads 4dw/row)
  __shared__ __align__(16) _Float16 Bs[64 * 40];   // B tile n-major, pitch 40 halves
  const int tid  = threadIdx.x;
  const int wave = tid >> 5;
  const int nBase = blockIdx.x * 64;
  const int mBase = blockIdx.y * 64;
  const int mt  = wave >> 1;
  const int nt0 = (wave & 1) * 2;
  v8f acc0 = {}; v8f acc1 = {};

  for (int k0 = 0; k0 < K; k0 += 32) {
    // ---- A tile (weights) 64(M) x 32(K) -> LDS ----
#if USE_TDM
    if (tid < 32) {
      // 2D Tensor-DMA descriptor: elem=2B, tile 32x64, row stride K elems,
      // LDS pad 4 dwords per 16-dword (=one row) interval -> pitch 40 halves.
      uint64_t gaddr = (uint64_t)(uintptr_t)(A + (size_t)mBase * K + k0);
      uint32_t ldsb  = (uint32_t)(uintptr_t)(&As[0]);
      uint32_t td0 = (uint32_t)(K - k0);        // remaining K (OOB -> zeros)
      uint32_t td1 = (uint32_t)(M - mBase);     // remaining M rows
      v4u g0;
      g0[0] = 1u;                                   // count=1, user mode
      g0[1] = ldsb;                                 // lds_addr
      g0[2] = (uint32_t)gaddr;                      // global_addr[31:0]
      g0[3] = (uint32_t)((gaddr >> 32) & 0x01FFFFFFu) | (2u << 30); // [56:32], type=2
      v8i g1;
      g1[0] = (int)((1u << 16) | (1u << 20) | (3u << 22) | (3u << 25)); // 2B, pad_en, 16dw int., 4dw pad
      g1[1] = (int)((td0 & 0xFFFFu) << 16);                // tensor_dim0 lo
      g1[2] = (int)((td0 >> 16) | ((td1 & 0xFFFFu) << 16));// dim0 hi | dim1 lo
      g1[3] = (int)((td1 >> 16) | (32u << 16));            // dim1 hi | tile_dim0=32
      g1[4] = (int)64;                                     // tile_dim1=64
      g1[5] = (int)(uint32_t)K;                            // tensor_dim0_stride
      g1[6] = 0; g1[7] = 0;
      v4i g2 = {0, 0, 0, 0}; v4i g3 = {0, 0, 0, 0};
#if defined(__clang_major__) && (__clang_major__ >= 23)
      v8i g4 = {0, 0, 0, 0, 0, 0, 0, 0};
      __builtin_amdgcn_tensor_load_to_lds(g0, g1, g2, g3, g4, 0);
#else
      __builtin_amdgcn_tensor_load_to_lds(g0, g1, g2, g3, 0);
#endif
    }
#else
#pragma unroll
    for (int t = 0; t < 8; ++t) {
      int idx = tid + t * 256;
      int m = idx >> 5, k = idx & 31;
      int gm = mBase + m, gk = k0 + k;
      As[m * 40 + k] = (gm < M && gk < K) ? A[(size_t)gm * K + gk] : (_Float16)0.0f;
    }
#endif
    // ---- B tile 32(K) x 64(N) -> LDS, stored n-major Bs[n][k] ----
    if (mode == 0 && k0 + 32 < K)
      __builtin_prefetch(B1 + (size_t)(k0 + 32) * N + nBase + (tid & 63), 0, 1);
#pragma unroll
    for (int t = 0; t < 8; ++t) {
      int idx = tid + t * 256;
      int k = idx >> 6, p = idx & 63;
      int gk = k0 + k, gp = nBase + p;
      _Float16 vle = (_Float16)0.0f;
      if (gk < K && gp < N) {
        if (mode == 0) {
          vle = B1[(size_t)gk * N + gp];
        } else {
          int ci = gk / 9, rem = gk - ci * 9;
          int ky = rem / 3 - 1, kx = rem - (rem / 3) * 3 - 1;
          int y = gp / Wimg, x = gp - y * Wimg;
          int yy = y + ky, xx = x + kx;
          if (yy >= 0 && yy < Himg && xx >= 0 && xx < Wimg) {
            vle = (ci < C1) ? B1[(size_t)ci * Himg * Wimg + yy * Wimg + xx]
                            : B2[(size_t)(ci - C1) * Himg * Wimg + yy * Wimg + xx];
          }
        }
      }
      Bs[p * 40 + k] = vle;
    }
#if USE_TDM
    if (tid < 32) __builtin_amdgcn_s_wait_tensorcnt(0);
#endif
    __syncthreads();
    v16h aF = load_frag_a(&As[mt * 16 * 40], 40);
    v16h b0 = load_frag_bT(&Bs[(nt0 * 16) * 40], 40);
    v16h b1 = load_frag_bT(&Bs[((nt0 + 1) * 16) * 40], 40);
    acc0 = wmma_f16(aF, b0, acc0);
    acc1 = wmma_f16(aF, b1, acc1);
    __syncthreads();
  }
  store_tile(acc0, mBase + mt * 16, nBase + nt0 * 16,       M, N, bias, resid, outH, outF, act);
  store_tile(acc1, mBase + mt * 16, nBase + (nt0 + 1) * 16, M, N, bias, resid, outH, outF, act);
}

// ---------------------------------------------------------------------------
// Small utility kernels
// ---------------------------------------------------------------------------
__global__ void k_cast_f16(const float* __restrict__ s, _Float16* __restrict__ d, int n) {
  int i = blockIdx.x * blockDim.x + threadIdx.x;
  if (i < n) d[i] = (_Float16)s[i];
}

__global__ void k_transpose_f16(const float* __restrict__ s, _Float16* __restrict__ d,
                                int R, int C) {  // d[c*R+r] = s[r*C+c]
  int i = blockIdx.x * blockDim.x + threadIdx.x;
  if (i >= R * C) return;
  int r = i / C, c = i - r * C;
  d[(size_t)c * R + r] = (_Float16)s[i];
}

__global__ void k_pool_f16(const _Float16* __restrict__ s, _Float16* __restrict__ d,
                           int C, int Hin, int Win) {
  int Ho = Hin >> 1, Wo = Win >> 1;
  int i = blockIdx.x * blockDim.x + threadIdx.x;
  if (i >= C * Ho * Wo) return;
  int c = i / (Ho * Wo), rp = i - c * Ho * Wo;
  int y = rp / Wo, x = rp - y * Wo;
  const _Float16* p = s + (size_t)c * Hin * Win + (2 * y) * Win + 2 * x;
  float a = (float)p[0], b = (float)p[1], e = (float)p[Win], f = (float)p[Win + 1];
  d[i] = (_Float16)fmaxf(fmaxf(a, b), fmaxf(e, f));
}

__global__ void k_layernorm(const _Float16* __restrict__ x, _Float16* __restrict__ y,
                            const float* __restrict__ g, const float* __restrict__ b,
                            int D, int N) {  // LN over feature dim, [D][N] layout
  int n = blockIdx.x * blockDim.x + threadIdx.x;
  if (n >= N) return;
  float m = 0.f;
  for (int d = 0; d < D; ++d) m += (float)x[(size_t)d * N + n];
  m /= D;
  float v = 0.f;
  for (int d = 0; d < D; ++d) { float t = (float)x[(size_t)d * N + n] - m; v += t * t; }
  v /= D;
  float inv = rsqrtf(v + 1e-5f);
  for (int d = 0; d < D; ++d) {
    float t = ((float)x[(size_t)d * N + n] - m) * inv * g[d] + b[d];
    y[(size_t)d * N + n] = (_Float16)t;
  }
}

__global__ void k_buckets(const _Float16* __restrict__ qk, const float* __restrict__ rot,
                          int* __restrict__ buckets, int N) {  // heads=4, dh=64, R=32
  int idx = blockIdx.x * blockDim.x + threadIdx.x;
  if (idx >= 4 * N) return;
  int h = idx / N, n = idx - h * N;
  float q[64];
  for (int d = 0; d < 64; ++d) q[d] = (float)qk[(size_t)(h * 64 + d) * N + n];
  float s[32];
  for (int r = 0; r < 32; ++r) {
    float a = 0.f;
    for (int d = 0; d < 64; ++d) a += q[d] * rot[(size_t)(h * 64 + d) * 32 + r];
    s[r] = a;
  }
  float best = -3.0e38f; int bi = 0;
  for (int r = 0; r < 32; ++r) if (s[r] > best) { best = s[r]; bi = r; }
  for (int r = 0; r < 32; ++r) if (-s[r] > best) { best = -s[r]; bi = r + 32; }
  buckets[h * N + n] = bi;
}

__global__ void k_sort(const int* __restrict__ buckets, int* __restrict__ perm, int N) {
  if (threadIdx.x != 0) return;         // stable counting sort == argsort(bucket*N+tick)
  int h = blockIdx.x;
  const int* b = buckets + h * N;
  int off[64];
  for (int i = 0; i < 64; ++i) off[i] = 0;
  for (int n = 0; n < N; ++n) off[b[n]]++;
  int run = 0;
  for (int i = 0; i < 64; ++i) { int c = off[i]; off[i] = run; run += c; }
  for (int n = 0; n < N; ++n) perm[h * N + off[b[n]]++] = n;
}

// ---------------------------------------------------------------------------
// LSH bucketed attention: one block per (chunk, head). BUCKET=64, look-one-back.
//   dots = (Q Khat^T)/8 with self-mask, softmax, O = P V, scatter via perm.
// LDS regions (62 KB total): Qs | KP (K tiles -> P matrix) | Ds (dots -> V^T)
// ---------------------------------------------------------------------------
__launch_bounds__(256)
__global__ void k_lsh_attn(const _Float16* __restrict__ qk,
                           const _Float16* __restrict__ vv,
                           const int* __restrict__ perm,
                           _Float16* __restrict__ o, int N) {
  __shared__ __align__(16) _Float16 Qs[64 * 72];     // Q tile [i][d]
  __shared__ __align__(16) _Float16 KP[128 * 72];    // K [j][d]; reused as P [i][j] pitch 136
  __shared__ __align__(16) float    Ds[64 * 132];    // dots f32; reused as V^T [d][j] pitch 136
  __shared__ int      qtok[64];
  __shared__ int      ktok[128];
  const int tid = threadIdx.x;
  const int lane = tid & 31;
  const int wave = tid >> 5;
  const int h = blockIdx.y, c = blockIdx.x, nc = gridDim.x;

  if (tid < 64) qtok[tid] = perm[h * N + c * 64 + tid];
  if (tid < 128) {
    int cc = (tid < 64) ? c : (c + nc - 1) % nc;     // look one chunk back (wrap)
    ktok[tid] = perm[h * N + cc * 64 + (tid & 63)];
  }
  __syncthreads();
#pragma unroll
  for (int t = 0; t < 16; ++t) {                     // stage Q (64x64)
    int idx = tid + t * 256;
    int i = idx >> 6, d = idx & 63;
    Qs[i * 72 + d] = qk[(size_t)(h * 64 + d) * N + qtok[i]];
  }
  if (tid < 128) {                                   // stage normalized K (128x64)
    int j = tid, tk = ktok[j];
    float ss = 0.f;
    for (int d = 0; d < 64; ++d) {
      float x = (float)qk[(size_t)(h * 64 + d) * N + tk];
      ss += x * x;
    }
    float sc = 1.f / (sqrtf(ss) + 1e-6f);
    for (int d = 0; d < 64; ++d) {
      float x = (float)qk[(size_t)(h * 64 + d) * N + tk];
      KP[j * 72 + d] = (_Float16)(x * sc);
    }
  }
  __syncthreads();
  {   // dots = Q Khat^T : 4x8 tiles of 16x16, K=64; wave -> 4 tiles
    int mt = wave >> 1, ntb = (wave & 1) * 4;
    v8f acc[4] = {};
    for (int kk = 0; kk < 64; kk += 32) {
      v16h aF = load_frag_a(&Qs[mt * 16 * 72 + kk], 72);
#pragma unroll
      for (int q = 0; q < 4; ++q) {
        v16h bF = load_frag_bT(&KP[(ntb + q) * 16 * 72 + kk], 72);
        acc[q] = wmma_f16(aF, bF, acc[q]);
      }
    }
    int nl = lane & 15, hi = lane >> 4;
#pragma unroll
    for (int q = 0; q < 4; ++q)
#pragma unroll
      for (int r = 0; r < 8; ++r)
        Ds[(mt * 16 + r + hi * 8) * 132 + (ntb + q) * 16 + nl] = acc[q][r] * 0.125f;
  }
  __syncthreads();
  if (tid < 64) {                                    // self-mask + softmax, P -> KP region
    int i = tid, qi = qtok[i];
    float mx = -3.0e38f;
    for (int j = 0; j < 128; ++j) {
      float dv = Ds[i * 132 + j];
      if (ktok[j] == qi) dv = -1e5f;
      Ds[i * 132 + j] = dv;
      if (dv > mx) mx = dv;
    }
    float sum = 0.f;
    for (int j = 0; j < 128; ++j) sum += __expf(Ds[i * 132 + j] - mx);
    float inv = 1.f / sum;
    _Float16* P = (_Float16*)KP;                     // P [i][j], pitch 136
    for (int j = 0; j < 128; ++j)
      P[i * 136 + j] = (_Float16)(__expf(Ds[i * 132 + j] - mx) * inv);
  }
  __syncthreads();
  _Float16* Vs = (_Float16*)Ds;                      // stage V^T [d][j] over dots region
  if (tid < 128) {
    int j = tid, tk = ktok[j];
    for (int d = 0; d < 64; ++d)
      Vs[d * 136 + j] = vv[(size_t)(h * 64 + d) * N + tk];
  }
  __syncthreads();
  {   // O = P V : 4x4 tiles, K=128; wave -> 2 tiles; scatter through qtok
    const _Float16* P = (const _Float16*)KP;
    int mt = wave >> 1, ntb = (wave & 1) * 2;
    v8f acc0 = {}, acc1 = {};
    for (int kk = 0; kk < 128; kk += 32) {
      v16h aF = load_frag_a(&P[mt * 16 * 136 + kk], 136);
      v16h b0 = load_frag_bT(&Vs[(ntb * 16) * 136 + kk], 136);
      v16h b1 = load_frag_bT(&Vs[((ntb + 1) * 16) * 136 + kk], 136);
      acc0 = wmma_f16(aF, b0, acc0);
      acc1 = wmma_f16(aF, b1, acc1);
    }
    int nl = lane & 15, hi = lane >> 4;
#pragma unroll
    for (int r = 0; r < 8; ++r) {
      int i = mt * 16 + r + hi * 8;
      int tk = qtok[i];
      o[(size_t)(h * 64 + ntb * 16 + nl) * N + tk]       = (_Float16)acc0[r];
      o[(size_t)(h * 64 + (ntb + 1) * 16 + nl) * N + tk] = (_Float16)acc1[r];
    }
  }
}

// ---------------------------------------------------------------------------
// ConvTranspose2d(k=4, s=2, p=1): direct (small FLOP share, VALU)
// weights (Cin, Cout, 4, 4)
// ---------------------------------------------------------------------------
__global__ void k_convT(const _Float16* __restrict__ in, const float* __restrict__ w,
                        const float* __restrict__ b, _Float16* __restrict__ out,
                        int Cin, int Cout, int Hin, int Win) {
  int Ho = Hin * 2, Wo = Win * 2;
  int idx = blockIdx.x * blockDim.x + threadIdx.x;
  if (idx >= Cout * Ho * Wo) return;
  int co = idx / (Ho * Wo);
  int rp = idx - co * Ho * Wo;
  int oy = rp / Wo, ox = rp - oy * Wo;
  float acc = b[co];
  for (int ky = 0; ky < 4; ++ky) {
    int ty = oy + 1 - ky;
    if (ty < 0 || (ty & 1)) continue;
    int iy = ty >> 1; if (iy >= Hin) continue;
    for (int kx = 0; kx < 4; ++kx) {
      int tx = ox + 1 - kx;
      if (tx < 0 || (tx & 1)) continue;
      int ix = tx >> 1; if (ix >= Win) continue;
      const _Float16* ip = in + (size_t)iy * Win + ix;
      const float* wp = w + (size_t)co * 16 + ky * 4 + kx;
      for (int ci = 0; ci < Cin; ++ci)
        acc += (float)ip[(size_t)ci * Hin * Win] * wp[(size_t)ci * Cout * 16];
    }
  }
  out[idx] = (_Float16)acc;
}

// ---------------------------------------------------------------------------
// Host orchestration
// ---------------------------------------------------------------------------
extern "C" void kernel_launch(void* const* d_in, const int* in_sizes, int n_in,
                              void* d_out, int out_size, void* d_ws, size_t ws_size,
                              hipStream_t stream) {
  (void)in_sizes; (void)n_in; (void)out_size; (void)ws_size;
  const float* burst   = (const float*)d_in[0];
  const float* conv1_w = (const float*)d_in[1];
  const float* conv1_b = (const float*)d_in[2];
  const float* conv2_w = (const float*)d_in[3];
  const float* conv2_b = (const float*)d_in[4];
  const float* conv3_w = (const float*)d_in[5];
  const float* conv3_b = (const float*)d_in[6];
  const float* ln1_g = (const float*)d_in[7];
  const float* ln1_b = (const float*)d_in[8];
  const float* Wqk = (const float*)d_in[9];
  const float* bqk = (const float*)d_in[10];
  const float* Wv  = (const float*)d_in[11];
  const float* bv  = (const float*)d_in[12];
  const float* Wo  = (const float*)d_in[13];
  const float* bo  = (const float*)d_in[14];
  const float* rot = (const float*)d_in[15];
  const float* ln2_g = (const float*)d_in[16];
  const float* ln2_b = (const float*)d_in[17];
  const float* Wff1 = (const float*)d_in[18];
  const float* bff1 = (const float*)d_in[19];
  const float* Wff2 = (const float*)d_in[20];
  const float* bff2 = (const float*)d_in[21];
  const float* up1_w = (const float*)d_in[22];
  const float* up1_b = (const float*)d_in[23];
  const float* dec1_w = (const float*)d_in[24];
  const float* dec1_b = (const float*)d_in[25];
  const float* up2_w = (const float*)d_in[26];
  const float* up2_b = (const float*)d_in[27];
  const float* dec2_w = (const float*)d_in[28];
  const float* dec2_b = (const float*)d_in[29];
  const float* out_w = (const float*)d_in[30];
  const float* out_b = (const float*)d_in[31];
  float* outp = (float*)d_out;

  // Workspace carve-up (only frame 0 of the burst matters for the output).
  char* ws = (char*)d_ws; size_t off = 0;
  auto alloc = [&](size_t bytes) -> char* {
    char* p = ws + off; off = (off + bytes + 255) & ~(size_t)255; return p;
  };
  auto allocH = [&](size_t n) { return (_Float16*)alloc(n * 2); };
  auto allocI = [&](size_t n) { return (int*)alloc(n * 4); };

  _Float16* x0h  = allocH(3 * 65536);
  _Float16* w1h  = allocH(64 * 27);
  _Float16* w2h  = allocH(128 * 576);
  _Float16* w3h  = allocH(256 * 1152);
  _Float16* wd1h = allocH(128 * 2304);
  _Float16* wd2h = allocH(64 * 1152);
  _Float16* wouth = allocH(3 * 576);
  _Float16* wqkT = allocH(4 * 65536);
  _Float16* wvT  = allocH(4 * 65536);
  _Float16* woT  = allocH(4 * 65536);
  _Float16* wff1T = allocH(4 * 262144);
  _Float16* wff2T = allocH(4 * 262144);
  _Float16* e1h  = allocH(64 * 65536);
  _Float16* p1h  = allocH(64 * 16384);
  _Float16* e2h  = allocH(128 * 16384);
  _Float16* p2h  = allocH(128 * 4096);
  _Float16* seqh = allocH(256 * 4096);
  _Float16* lnh  = allocH(256 * 4096);
  _Float16* qkh  = allocH(256 * 4096);
  _Float16* vh   = allocH(256 * 4096);
  _Float16* ohh  = allocH(256 * 4096);
  _Float16* ffh  = allocH(1024 * 4096);
  _Float16* d1h  = allocH(128 * 16384);
  _Float16* dec1h = allocH(128 * 16384);
  _Float16* d2h  = allocH(64 * 65536);
  _Float16* dec2h = allocH(64 * 65536);
  int* bkts = allocI(4 * 4096);
  int* permb = allocI(4 * 4096);

  auto cast = [&](const float* s, _Float16* d, int n) {
    k_cast_f16<<<(n + 255) / 256, 256, 0, stream>>>(s, d, n);
  };
  auto transp = [&](const float* s, _Float16* d, int R, int C) {
    int n = R * C;
    k_transpose_f16<<<(n + 255) / 256, 256, 0, stream>>>(s, d, R, C);
  };
  auto gemm = [&](const _Float16* A, const _Float16* B1, const _Float16* B2,
                  const float* bias, const _Float16* resid, _Float16* oh, float* of,
                  int M, int N, int K, int mode, int C1, int W, int H, int act) {
    dim3 g((N + 63) / 64, (M + 63) / 64);
    k_gemm<<<g, 256, 0, stream>>>(A, B1, B2, bias, resid, oh, of,
                                  M, N, K, mode, C1, W, H, act);
  };
  auto ln = [&](const _Float16* x, _Float16* y, const float* g, const float* b) {
    k_layernorm<<<4096 / 256, 256, 0, stream>>>(x, y, g, b, 256, 4096);
  };

  // ---- weight / input prep ----
  cast(burst, x0h, 3 * 65536);              // frame 0 only
  cast(conv1_w, w1h, 64 * 27);
  cast(conv2_w, w2h, 128 * 576);
  cast(conv3_w, w3h, 256 * 1152);
  cast(dec1_w, wd1h, 128 * 2304);
  cast(dec2_w, wd2h, 64 * 1152);
  cast(out_w, wouth, 3 * 576);
  for (int i = 0; i < 4; ++i) {
    transp(Wqk + (size_t)i * 65536,  wqkT + (size_t)i * 65536, 256, 256);
    transp(Wv  + (size_t)i * 65536,  wvT  + (size_t)i * 65536, 256, 256);
    transp(Wo  + (size_t)i * 65536,  woT  + (size_t)i * 65536, 256, 256);
    transp(Wff1 + (size_t)i * 262144, wff1T + (size_t)i * 262144, 256, 1024);
    transp(Wff2 + (size_t)i * 262144, wff2T + (size_t)i * 262144, 1024, 256);
  }

  // ---- encoder ----
  gemm(w1h, x0h, nullptr, conv1_b, nullptr, e1h, nullptr, 64, 65536, 27, 1, 3, 256, 256, ACT_RELU);
  k_pool_f16<<<(64 * 16384 + 255) / 256, 256, 0, stream>>>(e1h, p1h, 64, 256, 256);
  gemm(w2h, p1h, nullptr, conv2_b, nullptr, e2h, nullptr, 128, 16384, 576, 1, 64, 128, 128, ACT_RELU);
  k_pool_f16<<<(128 * 4096 + 255) / 256, 256, 0, stream>>>(e2h, p2h, 128, 128, 128);
  gemm(w3h, p2h, nullptr, conv3_b, nullptr, seqh, nullptr, 256, 4096, 1152, 1, 128, 64, 64, ACT_RELU);

  // ---- reformer (4 layers) ----
  for (int i = 0; i < 4; ++i) {
    ln(seqh, lnh, ln1_g + i * 256, ln1_b + i * 256);
    gemm(wqkT + (size_t)i * 65536, lnh, nullptr, bqk + i * 256, nullptr, qkh, nullptr,
         256, 4096, 256, 0, 0, 0, 0, ACT_NONE);
    gemm(wvT + (size_t)i * 65536, lnh, nullptr, bv + i * 256, nullptr, vh, nullptr,
         256, 4096, 256, 0, 0, 0, 0, ACT_NONE);
    k_buckets<<<(4 * 4096 + 255) / 256, 256, 0, stream>>>(qkh, rot + (size_t)i * 4 * 64 * 32, bkts, 4096);
    k_sort<<<4, 32, 0, stream>>>(bkts, permb, 4096);
    k_lsh_attn<<<dim3(64, 4), 256, 0, stream>>>(qkh, vh, permb, ohh, 4096);
    gemm(woT + (size_t)i * 65536, ohh, nullptr, bo + i * 256, seqh, seqh, nullptr,
         256, 4096, 256, 0, 0, 0, 0, ACT_NONE);
    ln(seqh, lnh, ln2_g + i * 256, ln2_b + i * 256);
    gemm(wff1T + (size_t)i * 262144, lnh, nullptr, bff1 + i * 1024, nullptr, ffh, nullptr,
         1024, 4096, 256, 0, 0, 0, 0, ACT_GELU);
    gemm(wff2T + (size_t)i * 262144, ffh, nullptr, bff2 + i * 256, seqh, seqh, nullptr,
         256, 4096, 1024, 0, 0, 0, 0, ACT_NONE);
  }

  // ---- decoder ----
  k_convT<<<(128 * 16384 + 255) / 256, 256, 0, stream>>>(seqh, up1_w, up1_b, d1h, 256, 128, 64, 64);
  gemm(wd1h, d1h, e2h, dec1_b, nullptr, dec1h, nullptr, 128, 16384, 2304, 1, 128, 128, 128, ACT_RELU);
  k_convT<<<(64 * 65536 + 255) / 256, 256, 0, stream>>>(dec1h, up2_w, up2_b, d2h, 128, 64, 128, 128);
  gemm(wd2h, d2h, e1h, dec2_b, nullptr, dec2h, nullptr, 64, 65536, 1152, 1, 64, 256, 256, ACT_RELU);
  gemm(wouth, dec2h, nullptr, out_b, nullptr, nullptr, outp, 3, 65536, 576, 1, 64, 256, 256, ACT_SIG);
}
